// LengthRegulator_50474455662964
// MI455X (gfx1250) — matrix operand order repaired and checked by reading, not compiled, not run on hardware
//
#include <hip/hip_runtime.h>
#include <hip/hip_bf16.h>

typedef __attribute__((ext_vector_type(16))) _Float16 v16h;
typedef __attribute__((ext_vector_type(8)))  _Float16 v8h;
typedef __attribute__((ext_vector_type(4)))  _Float16 v4h;
typedef __attribute__((ext_vector_type(8)))  float    v8f;

#define BB    16
#define NN    512
#define DD    256
#define TT    3072
#define KSTEPS 24          // 768 / 32
#define ROWS   32          // M rows per block

// ---------------------------------------------------------------------------
// Fused conv1d(K=3, SAME) -> +bias -> LayerNorm -> ReLU   (one block = 32 rows,
// all 256 output channels so LN can be done in-block).
// FIRST=true : input is f32 x, writes fp16 h1.
// FIRST=false: input is fp16 h1, additionally fuses the 256->1 linear + ReLU
//              and writes dpo directly.
// GEMM: [32 x 768] x [768 x 256], f16 WMMA 16x16x32, f32 accumulate.
// ---------------------------------------------------------------------------
template <bool FIRST>
__global__ __launch_bounds__(256)
void conv_ln_kernel(const void* __restrict__ in_v,
                    const float* __restrict__ w,      // [3][256][256]
                    const float* __restrict__ bias,   // [256]
                    const float* __restrict__ g,      // [256]
                    const float* __restrict__ bp,     // [256]
                    _Float16* __restrict__ out_h,     // h1 (FIRST)
                    const float* __restrict__ lw,     // [256] (!FIRST)
                    const float* __restrict__ lb,     // [1]   (!FIRST)
                    float* __restrict__ dpo)          // [B*N] (!FIRST)
{
    __shared__ alignas(16) _Float16 aT[ROWS * 32];     // A tile  (rows x k)
    __shared__ alignas(16) _Float16 bT[256 * 32];      // B tile transposed (col x k)
    __shared__ float cst[ROWS * 256];                  // C staging for LN
    __shared__ float mu_s[ROWS], rs_s[ROWS];

    const int tid  = threadIdx.x;
    const int wave = tid >> 5;            // 8 waves
    const int lane = tid & 31;            // wave32
    const int lrow = lane & 15;
    const int hi   = (lane >= 16) ? 1 : 0;
    const int m0   = blockIdx.x * ROWS;   // 512 % ROWS == 0 -> no batch crossing
    const int colbase = wave * 32;        // this wave's 32 output channels

    // Per-thread A-staging geometry: one contiguous 4-element chunk per step.
    // 32 rows x 32 k = 1024 elements = 256 threads x 4.
    const int a_r  = tid >> 3;            // 0..31  (tile row)
    const int a_k0 = (tid & 7) * 4;       // 0,4,..,28 (k offset within step)
    const int a_m  = m0 + a_r;
    const int a_b  = a_m >> 9;
    const int a_nb = (a_m & 511) - 1;     // halo row for kslot=0; add kslot later

    v8f acc[2][2];
    #pragma unroll
    for (int i = 0; i < 2; ++i)
        #pragma unroll
        for (int j = 0; j < 2; ++j) acc[i][j] = (v8f){};

    for (int step = 0; step < KSTEPS; ++step) {
        const int kslot = step >> 3;           // which conv tap (0..2)
        const int ci0   = (step & 7) * 32;     // input-channel base
        __syncthreads();

        // ---- stage A tile (ROWS x 32) as f16: one vector load, one predicate ----
        {
            const int n = a_nb + kslot;
            v4h hv = {};
            if (n >= 0 && n < NN) {
                const int idx = (a_b * NN + n) * DD + ci0 + a_k0;
                if (FIRST) {
                    const float4 f = *(const float4*)&((const float*)in_v)[idx];
                    hv[0] = (_Float16)f.x; hv[1] = (_Float16)f.y;
                    hv[2] = (_Float16)f.z; hv[3] = (_Float16)f.w;
                } else {
                    hv = *(const v4h*)&((const _Float16*)in_v)[idx];
                }
            }
            *(v4h*)&aT[a_r * 32 + a_k0] = hv;
        }
        // ---- stage B tile transposed: bT[co*32 + k] = w[kslot][ci0+k][co] ----
        {
            const float* wp = w + kslot * (DD * 256) + ci0 * 256;
            #pragma unroll
            for (int j = 0; j < 32; ++j)
                bT[tid * 32 + j] = (_Float16)wp[j * 256 + tid];
            // prefetch next step's weight slab into cache (global_prefetch_b8)
            if (step + 1 < KSTEPS) {
                const int ks2 = (step + 1) >> 3;
                const int ci2 = ((step + 1) & 7) * 32;
                __builtin_prefetch(w + ks2 * (DD * 256) + ci2 * 256 + tid, 0, 1);
            }
        }
        __syncthreads();

        // ---- load fragments and issue WMMAs ----
        // A frag (16x32 f16): lane half selects K groups {0-7,16-23} / {8-15,24-31}
        v16h af[2];
        #pragma unroll
        for (int rt = 0; rt < 2; ++rt) {
            const int row = rt * 16 + lrow;
            const int kb  = hi ? 8 : 0;
            const v8h lo = *(const v8h*)&aT[row * 32 + kb];
            const v8h hv = *(const v8h*)&aT[row * 32 + kb + 16];
            #pragma unroll
            for (int i = 0; i < 8; ++i) { af[rt][i] = lo[i]; af[rt][8 + i] = hv[i]; }
        }
        // B frag (32x16 f16): lane half selects K groups {0-7,16-23} / {8-15,24-31}
        v16h bf[2];
        #pragma unroll
        for (int cj = 0; cj < 2; ++cj) {
            const int col = colbase + cj * 16 + lrow;
            const int kb  = hi ? 8 : 0;
            const v8h lo = *(const v8h*)&bT[col * 32 + kb];
            const v8h hv = *(const v8h*)&bT[col * 32 + kb + 16];
            #pragma unroll
            for (int i = 0; i < 8; ++i) { bf[cj][i] = lo[i]; bf[cj][8 + i] = hv[i]; }
        }
        #pragma unroll
        for (int rt = 0; rt < 2; ++rt)
            #pragma unroll
            for (int cj = 0; cj < 2; ++cj)
                acc[rt][cj] = __builtin_amdgcn_wmma_f32_16x16x32_f16(
                    false, af[rt], false, bf[cj], (short)0, acc[rt][cj], false, false);
    }

    // ---- writeback accumulators (+bias) to LDS staging ----
    __syncthreads();
    {
        const int c0 = colbase + lrow, c1 = c0 + 16;
        const float bb0 = bias[c0], bb1 = bias[c1];
        #pragma unroll
        for (int rt = 0; rt < 2; ++rt) {
            #pragma unroll
            for (int v = 0; v < 8; ++v) {
                const int r = rt * 16 + v + (hi ? 8 : 0);
                cst[r * 256 + c0] = acc[rt][0][v] + bb0;
                cst[r * 256 + c1] = acc[rt][1][v] + bb1;
            }
        }
    }
    __syncthreads();

    // ---- LayerNorm stats (one thread per row; tiny vs GEMM) ----
    if (tid < ROWS) {
        float s = 0.f;
        for (int c = 0; c < 256; ++c) s += cst[tid * 256 + c];
        const float mu = s * (1.f / 256.f);
        float vs = 0.f;
        for (int c = 0; c < 256; ++c) { const float d = cst[tid * 256 + c] - mu; vs += d * d; }
        mu_s[tid] = mu;
        rs_s[tid] = rsqrtf(vs * (1.f / 256.f) + 1e-5f);
    }
    __syncthreads();

    const float gg  = g[tid];
    const float bpv = bp[tid];
    if (FIRST) {
        // normalize + relu, store fp16 activation (column tid, all rows)
        for (int i = 0; i < ROWS; ++i) {
            float v = (cst[i * 256 + tid] - mu_s[i]) * rs_s[i] * gg + bpv;
            v = v > 0.f ? v : 0.f;
            out_h[(m0 + i) * DD + tid] = (_Float16)v;
        }
    } else {
        // normalize + relu in place, then fused linear(256->1) + relu -> dpo
        for (int i = 0; i < ROWS; ++i) {
            float v = (cst[i * 256 + tid] - mu_s[i]) * rs_s[i] * gg + bpv;
            cst[i * 256 + tid] = v > 0.f ? v : 0.f;
        }
        __syncthreads();
        if (tid < ROWS) {
            float s = 0.f;
            for (int c = 0; c < 256; ++c) s += cst[tid * 256 + c] * lw[c];
            s += lb[0];
            dpo[m0 + tid] = s > 0.f ? s : 0.f;
        }
    }
}

// ---------------------------------------------------------------------------
// Length regulator step 1: per-batch inclusive scan of durations, then build
// the frame->token map in LDS (disjoint intervals -> race-free scatter).
// ---------------------------------------------------------------------------
__global__ __launch_bounds__(512)
void lr_map_kernel(const int* __restrict__ dur, int* __restrict__ map)
{
    __shared__ int s[NN];
    __shared__ int ms[TT];
    const int b = blockIdx.x, t = threadIdx.x;

    const int d = dur[b * NN + t];
    s[t] = d;
    for (int i = t; i < TT; i += 512) ms[i] = -1;
    __syncthreads();

    // Hillis-Steele inclusive scan over 512 entries
    for (int off = 1; off < NN; off <<= 1) {
        const int v = (t >= off) ? s[t - off] : 0;
        __syncthreads();
        s[t] += v;
        __syncthreads();
    }
    int end = s[t];
    const int start = end - d;
    if (end > TT) end = TT;
    for (int i = start; i < end; ++i) ms[i] = t;   // intervals disjoint
    __syncthreads();
    for (int i = t; i < TT; i += 512) map[b * TT + i] = ms[i];
}

// ---------------------------------------------------------------------------
// Length regulator step 2: gather rows of x (or zeros) into the output.
// One 64-lane group copies one 256-float row as float4; every output element
// written exactly once -> no pre-zeroing pass needed.
// ---------------------------------------------------------------------------
__global__ __launch_bounds__(256)
void lr_gather_kernel(const float* __restrict__ x,
                      const int* __restrict__ map,
                      float* __restrict__ out)
{
    const int gidx = blockIdx.x * 4 + (threadIdx.x >> 6);   // flat b*T + t
    const int l    = threadIdx.x & 63;
    const int b    = gidx / TT;
    const int n    = map[gidx];

    float4* dst = (float4*)out + (size_t)gidx * 64 + l;
    if (n >= 0) {
        const float4* src = (const float4*)x + ((size_t)(b * NN + n)) * 64 + l;
        *dst = *src;
    } else {
        *dst = make_float4(0.f, 0.f, 0.f, 0.f);
    }
}

// ---------------------------------------------------------------------------
extern "C" void kernel_launch(void* const* d_in, const int* in_sizes, int n_in,
                              void* d_out, int out_size, void* d_ws, size_t ws_size,
                              hipStream_t stream)
{
    (void)in_sizes; (void)n_in; (void)out_size; (void)ws_size;

    const float* x   = (const float*)d_in[0];
    const int*   dur = (const int*)d_in[1];
    // d_in[2] = mel_max_length (compile-time 3072)
    const float* c1w = (const float*)d_in[3];
    const float* c1b = (const float*)d_in[4];
    const float* g1  = (const float*)d_in[5];
    const float* b1  = (const float*)d_in[6];
    const float* c2w = (const float*)d_in[7];
    const float* c2b = (const float*)d_in[8];
    const float* g2  = (const float*)d_in[9];
    const float* b2  = (const float*)d_in[10];
    const float* lw  = (const float*)d_in[11];
    const float* lb  = (const float*)d_in[12];

    float* out = (float*)d_out;                       // [B, T, D]
    float* dpo = out + (size_t)BB * TT * DD;          // [B, N]

    int*      map = (int*)d_ws;                                   // B*T ints (192 KB)
    _Float16* h1  = (_Float16*)((char*)d_ws + 256 * 1024);        // B*N*FILT fp16 (4 MB)

    const int M = BB * NN;                      // 8192 rows
    conv_ln_kernel<true ><<<M / ROWS, 256, 0, stream>>>(x,  c1w, c1b, g1, b1, h1,
                                                        nullptr, nullptr, nullptr);
    conv_ln_kernel<false><<<M / ROWS, 256, 0, stream>>>(h1, c2w, c2b, g2, b2, nullptr,
                                                        lw, lb, dpo);

    lr_map_kernel   <<<BB, 512, 0, stream>>>(dur, map);
    lr_gather_kernel<<<(BB * TT) / 4, 256, 0, stream>>>(x, map, out);
}
